// EMVD_25864293056765
// MI455X (gfx1250) — matrix operand order repaired and checked by reading, not compiled, not run on hardware
//
#include <hip/hip_runtime.h>
#include <hip/hip_bf16.h>

// ---------------------------------------------------------------------------
// EMVD fusion + denoise pipeline for gfx1250 (MI455X).
// Convs implemented as implicit GEMM over v_wmma_f32_16x16x32_f16.
// Activations staged in f16 NHWC with 1-pixel zero halo; weights pulled into
// LDS by the Tensor Data Mover (TENSOR_LOAD_TO_LDS + s_wait_tensorcnt).
// ---------------------------------------------------------------------------

typedef __attribute__((ext_vector_type(16))) _Float16 v16h;
typedef __attribute__((ext_vector_type(8)))  _Float16 half8;
typedef __attribute__((ext_vector_type(8)))  float    v8f;
typedef __attribute__((ext_vector_type(4)))  unsigned int uint4v;
typedef __attribute__((ext_vector_type(8)))  int      int8v;
typedef __attribute__((ext_vector_type(4)))  int      int4v;

#if __has_builtin(__builtin_amdgcn_tensor_load_to_lds)
#define EMVD_HAVE_TDM 1
#else
#define EMVD_HAVE_TDM 0
#endif

#define BATCH 4
#define HH    256
#define WW    256
#define HP    258
#define WP    258
#define PLANE ((size_t)HH * WW)   // 65536

// ---------------------------------------------------------------------------
// Weight repack: OIHW f32 -> [ky][kx][Opad][Cpad] f16 (zero padded)
// ---------------------------------------------------------------------------
__global__ void prep_weights(const float* __restrict__ src, _Float16* __restrict__ dst,
                             int O, int I, int Opad, int Cpad)
{
    int idx = blockIdx.x * blockDim.x + threadIdx.x;
    int total = 9 * Opad * Cpad;
    if (idx >= total) return;
    int c   = idx % Cpad;
    int o   = (idx / Cpad) % Opad;
    int pos = idx / (Cpad * Opad);
    int ky = pos / 3, kx = pos % 3;
    float v = 0.f;
    if (o < O && c < I)
        v = src[((o * I + c) * 3 + ky) * 3 + kx];
    dst[idx] = (_Float16)v;
}

// ---------------------------------------------------------------------------
// Build fusion_in (NHWC halo, Cpad=32): [ |ll1-ll0| x16, gamma_up x4, sigma_ll1 x4, 0 x8 ]
// Also emits sigma_ll0 / sigma_ll1 planar f32.
// ---------------------------------------------------------------------------
__global__ void prep_fusion_in(const float* __restrict__ ft0, const float* __restrict__ ft1,
                               const float* __restrict__ gup,
                               const float* __restrict__ ca, const float* __restrict__ cb,
                               _Float16* __restrict__ a0,
                               float* __restrict__ sig0, float* __restrict__ sig1)
{
    int idx = blockIdx.x * blockDim.x + threadIdx.x;
    if (idx >= BATCH * HP * WP) return;
    int xh = idx % WP;
    int yh = (idx / WP) % HP;
    int b  = idx / (WP * HP);
    _Float16* dst = a0 + (size_t)idx * 32;
    if (xh == 0 || xh == WP - 1 || yh == 0 || yh == HP - 1) {
        int4 z = make_int4(0, 0, 0, 0);
        #pragma unroll
        for (int i = 0; i < 4; ++i) ((int4*)dst)[i] = z;
        return;
    }
    int xx = xh - 1, yy = yh - 1;
    float A = *ca, Bc = *cb;
    __align__(16) _Float16 h[32];
    size_t p    = (size_t)yy * WW + xx;
    size_t b64  = (size_t)b * 64 * PLANE + p;
    size_t b4   = (size_t)b * 4 * PLANE + p;
    #pragma unroll
    for (int c = 0; c < 16; ++c) {
        float v0 = ft0[b64 + (size_t)c * PLANE];
        float v1 = ft1[b64 + (size_t)c * PLANE];
        h[c] = (_Float16)fabsf(v1 - v0);
        if (c < 4) {
            float s0 = fminf(fmaxf(v0, 0.f), 1.f) * A + Bc;
            float s1 = fminf(fmaxf(v1, 0.f), 1.f) * A + Bc;
            sig0[b4 + (size_t)c * PLANE] = s0;
            sig1[b4 + (size_t)c * PLANE] = s1;
            h[20 + c] = (_Float16)s1;
        }
    }
    #pragma unroll
    for (int c = 0; c < 4; ++c)
        h[16 + c] = (_Float16)gup[b4 + (size_t)c * PLANE];
    #pragma unroll
    for (int c = 24; c < 32; ++c) h[c] = (_Float16)0.f;
    #pragma unroll
    for (int i = 0; i < 4; ++i) ((int4*)dst)[i] = ((const int4*)h)[i];
}

// ---------------------------------------------------------------------------
// Zero the halo ring of the four [B][HP][WP][64] intermediate buffers.
// ---------------------------------------------------------------------------
__global__ void zero_halo4(_Float16* b0, _Float16* b1, _Float16* b2, _Float16* b3)
{
    int idx = blockIdx.x * blockDim.x + threadIdx.x;
    if (idx >= BATCH * HP * WP) return;
    int xh = idx % WP;
    int yh = (idx / WP) % HP;
    if (!(xh == 0 || xh == WP - 1 || yh == 0 || yh == HP - 1)) return;
    size_t off = (size_t)idx * 64;
    int4 z = make_int4(0, 0, 0, 0);
    _Float16* bufs[4] = {b0, b1, b2, b3};
    #pragma unroll
    for (int k = 0; k < 4; ++k) {
        int4* d = (int4*)(bufs[k] + off);
        #pragma unroll
        for (int i = 0; i < 8; ++i) d[i] = z;
    }
}

// ---------------------------------------------------------------------------
// Implicit-GEMM 3x3 conv via WMMA.
//   A (SRC0) = weight tile, 16 out-ch (M) x 32 K           -> from LDS
//   B (SRC1) = activation tile, 32 K x 16 pixels (N)       -> NHWC halo buffer
//   D lanes: lane(g,n) VGPR r -> (oc = base+8g+r, pixel = x0+n)
// ACT: 0 = relu -> f16 NHWC halo (Cpad = OTILES*16)
//      1 = sigmoid -> f32 gamma [B][4][H][W]   (OTILES must be 1)
//      2 = none -> f32 [B][64][H][W]
// ---------------------------------------------------------------------------
template <int CIN_PAD, int OTILES, int ACT>
__global__ __launch_bounds__(256)
void conv3x3_wmma(const _Float16* __restrict__ act,   // [B][HP][WP][CIN_PAD]
                  const _Float16* __restrict__ wgt,   // [9][OTILES*16][CIN_PAD]
                  const float* __restrict__ bias,
                  float* __restrict__ out_f32,
                  _Float16* __restrict__ out_f16)
{
    constexpr int OPAD = OTILES * 16;
    constexpr int WTOT = 9 * OPAD * CIN_PAD;                // halves
    constexpr unsigned ELEMS8 = (unsigned)(WTOT * 2 / 8);   // 8-byte elements

    extern __shared__ _Float16 ldsw[];

    const int lane = threadIdx.x & 31;
    const int wave = threadIdx.x >> 5;
    const int g = lane >> 4;      // half-wave group
    const int n = lane & 15;      // A row (out-ch in tile) == B col (pixel in tile)

    int bid = blockIdx.x;
    const int seg = bid & 1;  bid >>= 1;
    const int yy  = bid & 255; bid >>= 8;
    const int b   = bid;
    const int x0  = seg * 128 + wave * 16;    // interior x base for this wave

    const _Float16* actB = act + (size_t)b * HP * WP * CIN_PAD;

    // Prefetch this wave's three input rows while weights stage into LDS.
    {
        const char* r0 = (const char*)(actB + ((size_t)(yy + 0) * WP + x0) * CIN_PAD);
        const char* r1 = (const char*)(actB + ((size_t)(yy + 1) * WP + x0) * CIN_PAD);
        const char* r2 = (const char*)(actB + ((size_t)(yy + 2) * WP + x0) * CIN_PAD);
        __builtin_prefetch(r0 + lane * 128, 0, 1);
        __builtin_prefetch(r1 + lane * 128, 0, 1);
        __builtin_prefetch(r2 + lane * 128, 0, 1);
    }

#if EMVD_HAVE_TDM
    // Stage the whole [9][OPAD][CIN_PAD] f16 weight tensor into LDS with the
    // Tensor Data Mover: one 1-row tile of ELEMS8 8-byte elements.
    if (wave == 0) {
        unsigned long long ga = (unsigned long long)(uintptr_t)wgt;
        uint4v g0;
        g0[0] = 1u;                                   // count=1 (valid user D#)
        g0[1] = 0u;                                   // lds_addr = dynamic-LDS base
        g0[2] = (unsigned)ga;                         // global_addr[31:0]
        g0[3] = ((unsigned)(ga >> 32) & 0x01FFFFFFu)  // global_addr[56:32]
              | (2u << 30);                           // type = 2 ("image")
        int8v g1;
        g1[0] = (int)(3u << 16);                      // wg_mask=0, data_size=8B
        g1[1] = (int)((ELEMS8 & 0xFFFFu) << 16);      // tensor_dim0[15:0]
        g1[2] = (int)((ELEMS8 >> 16) | (1u << 16));   // tensor_dim0[31:16] | tensor_dim1=1
        g1[3] = (int)(ELEMS8 << 16);                  // tile_dim0 = ELEMS8 (<= 0xFFFF)
        g1[4] = 0;                                    // tile_dim1/2 unused
        g1[5] = (int)ELEMS8;                          // tensor_dim0_stride[31:0]
        g1[6] = 0;
        g1[7] = 0;
        int4v gz = {};
#if __clang_major__ >= 23
        int8v gz8 = {};
        __builtin_amdgcn_tensor_load_to_lds(g0, g1, gz, gz, gz8, 0);
#else
        __builtin_amdgcn_tensor_load_to_lds(g0, g1, gz, gz, 0);
#endif
        __builtin_amdgcn_s_wait_tensorcnt(0);
    }
#else
    {   // fallback: cooperative weight stage, 16B granules
        const int4* s = (const int4*)wgt;
        int4* d = (int4*)ldsw;
        for (int i = threadIdx.x; i < WTOT / 8; i += 256) d[i] = s[i];
    }
#endif
    __syncthreads();

    v8f acc[OTILES] = {};

    #pragma unroll
    for (int ky = 0; ky < 3; ++ky) {
        #pragma unroll
        for (int kx = 0; kx < 3; ++kx) {
            // halo coords: input pixel for interior (yy, x0+n) at tap (ky,kx)
            const _Float16* arow =
                actB + (((size_t)(yy + ky) * WP) + (size_t)(x0 + kx + n)) * CIN_PAD;
            const _Float16* wpos = ldsw + (size_t)(ky * 3 + kx) * OPAD * CIN_PAD;
            #pragma unroll
            for (int ch = 0; ch < CIN_PAD / 32; ++ch) {
                // B fragment: lane(g,n) halves i -> (K = 16g+i, pixel n): contiguous 32B
                v16h bfrag = *(const v16h*)(arow + ch * 32 + 16 * g);
                // Batch all A-fragment LDS loads so one wait covers the group
                v16h af[OTILES];
                #pragma unroll
                for (int t = 0; t < OTILES; ++t) {
                    // A fragment: lane(g,n) halves i -> K=16g+i
                    const _Float16* wrow = wpos + (size_t)(t * 16 + n) * CIN_PAD + ch * 32;
                    ((int4*)&af[t])[0] = *(const int4*)(wrow + 16 * g);
                    ((int4*)&af[t])[1] = *(const int4*)(wrow + 16 * g + 8);
                }
                #pragma unroll
                for (int t = 0; t < OTILES; ++t)
                    acc[t] = __builtin_amdgcn_wmma_f32_16x16x32_f16(
                        false, af[t], false, bfrag, (short)0, acc[t], false, false);
            }
        }
    }

    if constexpr (ACT == 0) {
        // relu -> f16 NHWC halo, Cpad = OPAD; lane stores 8 consecutive channels (16B)
        _Float16* orow =
            out_f16 + (((size_t)b * HP + (yy + 1)) * WP + (size_t)(x0 + 1 + n)) * OPAD;
        #pragma unroll
        for (int t = 0; t < OTILES; ++t) {
            half8 hv;
            #pragma unroll
            for (int r = 0; r < 8; ++r) {
                int oc = t * 16 + 8 * g + r;
                float v = acc[t][r] + bias[oc];
                v = v > 0.f ? v : 0.f;
                hv[r] = (_Float16)v;
            }
            *(half8*)(orow + t * 16 + 8 * g) = hv;
        }
    } else if constexpr (ACT == 1) {
        // sigmoid -> gamma f32 [B][4][H][W]; only oc<4 valid (g==0, r<4)
        if (g == 0) {
            #pragma unroll
            for (int r = 0; r < 4; ++r) {
                float v = acc[0][r] + bias[r];
                v = 1.f / (1.f + __expf(-v));
                out_f32[(((size_t)b * 4 + r) * HH + yy) * WW + (x0 + n)] = v;
            }
        }
    } else {
        // no activation -> f32 NCHW [B][64][H][W]; lanes n give consecutive x
        #pragma unroll
        for (int t = 0; t < OTILES; ++t) {
            #pragma unroll
            for (int r = 0; r < 8; ++r) {
                int oc = t * 16 + 8 * g + r;
                out_f32[(((size_t)b * 64 + oc) * HH + yy) * WW + (x0 + n)] =
                    acc[t][r] + bias[oc];
            }
        }
    }
}

// ---------------------------------------------------------------------------
// Fuse stage: gamma -> fusion_out, sigma, and denoise_in (NHWC halo, Cpad=128):
//   [ fusion_out x64, ll1 x16, denoise_down x16, sigma x4, 0 x28 ]
// ---------------------------------------------------------------------------
__global__ void fuse_kernel(const float* __restrict__ ft0, const float* __restrict__ ft1,
                            const float* __restrict__ dd,  const float* __restrict__ gamma,
                            const float* __restrict__ sig0, const float* __restrict__ sig1,
                            float* __restrict__ fusion_out, float* __restrict__ sigma_out,
                            _Float16* __restrict__ din)
{
    int idx = blockIdx.x * blockDim.x + threadIdx.x;
    if (idx >= BATCH * HP * WP) return;
    int xh = idx % WP;
    int yh = (idx / WP) % HP;
    int b  = idx / (WP * HP);
    _Float16* dst = din + (size_t)idx * 128;
    if (xh == 0 || xh == WP - 1 || yh == 0 || yh == HP - 1) {
        int4 z = make_int4(0, 0, 0, 0);
        #pragma unroll
        for (int i = 0; i < 16; ++i) ((int4*)dst)[i] = z;
        return;
    }
    int xx = xh - 1, yy = yh - 1;
    size_t p   = (size_t)yy * WW + xx;
    size_t b64 = (size_t)b * 64 * PLANE + p;
    size_t b16 = (size_t)b * 16 * PLANE + p;
    size_t b4  = (size_t)b * 4  * PLANE + p;

    __align__(16) _Float16 h[128];
    float gm[4];
    #pragma unroll
    for (int c = 0; c < 4; ++c) gm[c] = gamma[b4 + (size_t)c * PLANE];

    #pragma unroll 8
    for (int c = 0; c < 64; ++c) {
        float gv = gm[c >> 4];                 // jnp.repeat(gamma, 16, axis=1)
        float v0 = ft0[b64 + (size_t)c * PLANE];
        float v1 = ft1[b64 + (size_t)c * PLANE];
        float fo = v0 * (1.f - gv) + v1 * gv;
        fusion_out[b64 + (size_t)c * PLANE] = fo;
        h[c] = (_Float16)fo;
        if (c < 16) h[64 + c] = (_Float16)v1;  // ll1
    }
    #pragma unroll
    for (int c = 0; c < 16; ++c)
        h[80 + c] = (_Float16)dd[b16 + (size_t)c * PLANE];
    #pragma unroll
    for (int c = 0; c < 4; ++c) {
        float gv = gm[c];
        float s = (1.f - gv) * (1.f - gv) * sig0[b4 + (size_t)c * PLANE]
                + gv * gv * sig1[b4 + (size_t)c * PLANE];
        sigma_out[b4 + (size_t)c * PLANE] = s;
        h[96 + c] = (_Float16)s;
    }
    #pragma unroll
    for (int c = 100; c < 128; ++c) h[c] = (_Float16)0.f;
    #pragma unroll
    for (int i = 0; i < 16; ++i) ((int4*)dst)[i] = ((const int4*)h)[i];
}

// ---------------------------------------------------------------------------
// Host driver
// ---------------------------------------------------------------------------
extern "C" void kernel_launch(void* const* d_in, const int* in_sizes, int n_in,
                              void* d_out, int out_size, void* d_ws, size_t ws_size,
                              hipStream_t stream)
{
    const float* ft0 = (const float*)d_in[0];
    const float* ft1 = (const float*)d_in[1];
    const float* gup = (const float*)d_in[2];
    const float* dd  = (const float*)d_in[3];
    const float* ca  = (const float*)d_in[4];
    const float* cb  = (const float*)d_in[5];
    const float* fw1 = (const float*)d_in[6];  const float* fb1 = (const float*)d_in[7];
    const float* fw2 = (const float*)d_in[8];  const float* fb2 = (const float*)d_in[9];
    const float* fw3 = (const float*)d_in[10]; const float* fb3 = (const float*)d_in[11];
    const float* dw1 = (const float*)d_in[12]; const float* db1 = (const float*)d_in[13];
    const float* dw2 = (const float*)d_in[14]; const float* db2 = (const float*)d_in[15];
    const float* dw3 = (const float*)d_in[16]; const float* db3 = (const float*)d_in[17];

    float* out        = (float*)d_out;
    float* fusion_out = out;                               // 4*64*256*256
    float* denoise_o  = out + (size_t)4 * 64 * PLANE;      // 4*64*256*256
    float* gamma_o    = out + (size_t)8 * 64 * PLANE;      // 4*4*256*256
    float* sigma_o    = gamma_o + (size_t)4 * 4 * PLANE;   // 4*4*256*256

    // ---- workspace layout (256B aligned slabs) ----
    char* ws = (char*)d_ws;
    size_t off = 0;
    auto carve = [&](size_t bytes) {
        void* p = ws + off;
        off += (bytes + 255) & ~(size_t)255;
        return p;
    };
    const size_t HPWP = (size_t)HP * WP;
    _Float16* A0  = (_Float16*)carve(BATCH * HPWP * 32  * 2);
    _Float16* H1  = (_Float16*)carve(BATCH * HPWP * 64  * 2);
    _Float16* H2  = (_Float16*)carve(BATCH * HPWP * 64  * 2);
    _Float16* DIN = (_Float16*)carve(BATCH * HPWP * 128 * 2);
    _Float16* D1  = (_Float16*)carve(BATCH * HPWP * 64  * 2);
    _Float16* D2  = (_Float16*)carve(BATCH * HPWP * 64  * 2);
    float*    S0  = (float*)carve(BATCH * 4 * PLANE * 4);
    float*    S1  = (float*)carve(BATCH * 4 * PLANE * 4);
    _Float16* W1  = (_Float16*)carve((size_t)9 * 64 * 32  * 2);
    _Float16* W2  = (_Float16*)carve((size_t)9 * 64 * 64  * 2);
    _Float16* W3  = (_Float16*)carve((size_t)9 * 16 * 64  * 2);
    _Float16* DW1 = (_Float16*)carve((size_t)9 * 64 * 128 * 2);
    _Float16* DW2 = (_Float16*)carve((size_t)9 * 64 * 64  * 2);
    _Float16* DW3 = (_Float16*)carve((size_t)9 * 64 * 64  * 2);

    auto wgrid = [](int Opad, int Cpad) { return (9 * Opad * Cpad + 255) / 256; };
    prep_weights<<<wgrid(64, 32),  256, 0, stream>>>(fw1, W1,  64, 24,  64, 32);
    prep_weights<<<wgrid(64, 64),  256, 0, stream>>>(fw2, W2,  64, 64,  64, 64);
    prep_weights<<<wgrid(16, 64),  256, 0, stream>>>(fw3, W3,   4, 64,  16, 64);
    prep_weights<<<wgrid(64, 128), 256, 0, stream>>>(dw1, DW1, 64, 100, 64, 128);
    prep_weights<<<wgrid(64, 64),  256, 0, stream>>>(dw2, DW2, 64, 64,  64, 64);
    prep_weights<<<wgrid(64, 64),  256, 0, stream>>>(dw3, DW3, 64, 64,  64, 64);

    const int pixgrid = (BATCH * HP * WP + 255) / 256;
    prep_fusion_in<<<pixgrid, 256, 0, stream>>>(ft0, ft1, gup, ca, cb, A0, S0, S1);
    zero_halo4<<<pixgrid, 256, 0, stream>>>(H1, H2, D1, D2);

    // conv grid: B * H rows * 2 segments of 128 px; block = 8 waves x 16 px
    const int cgrid = BATCH * HH * 2;
    conv3x3_wmma<32, 4, 0><<<cgrid, 256, 9 * 64 * 32  * 2, stream>>>(A0,  W1,  fb1, nullptr, H1);
    conv3x3_wmma<64, 4, 0><<<cgrid, 256, 9 * 64 * 64  * 2, stream>>>(H1,  W2,  fb2, nullptr, H2);
    conv3x3_wmma<64, 1, 1><<<cgrid, 256, 9 * 16 * 64  * 2, stream>>>(H2,  W3,  fb3, gamma_o, nullptr);

    fuse_kernel<<<pixgrid, 256, 0, stream>>>(ft0, ft1, dd, gamma_o, S0, S1,
                                             fusion_out, sigma_o, DIN);

    conv3x3_wmma<128, 4, 0><<<cgrid, 256, 9 * 64 * 128 * 2, stream>>>(DIN, DW1, db1, nullptr, D1);
    conv3x3_wmma<64,  4, 0><<<cgrid, 256, 9 * 64 * 64  * 2, stream>>>(D1,  DW2, db2, nullptr, D2);
    conv3x3_wmma<64,  4, 2><<<cgrid, 256, 9 * 64 * 64  * 2, stream>>>(D2,  DW3, db3, denoise_o, nullptr);
}
